// SonataFlowV3_11647951306849
// MI455X (gfx1250) — matrix operand order repaired and checked by reading, not compiled, not run on hardware
//
#include <hip/hip_runtime.h>

#define B_ 2
#define T_ 2048
#define MEL_ 100
#define D_ 1024
#define H_ 16
#define L_ 6
#define DH_ 64
#define WIN_ 512
#define FF_ 4096
#define CD_ 1024

typedef __attribute__((ext_vector_type(16))) __bf16        v16bf;
typedef __attribute__((ext_vector_type(8)))  float         v8f;
typedef __attribute__((ext_vector_type(8)))  unsigned int  v8u;

union FragU { v8u u; v16bf b; uint4 q[2]; };

__device__ __forceinline__ unsigned short f2bf(float f) {
  unsigned int u = __float_as_uint(f);
  u += 0x7FFFu + ((u >> 16) & 1u);
  return (unsigned short)(u >> 16);
}
__device__ __forceinline__ float sigmoidf_(float x) { return 1.0f / (1.0f + __expf(-x)); }

// ---------------------------------------------------------------------------
// Weight fp32 [K x N] -> bf16 transposed/padded [Npad x K]
// ---------------------------------------------------------------------------
__global__ __launch_bounds__(256) void wt_convert(
    const float* __restrict__ W, unsigned short* __restrict__ Wt,
    int K, int kshift, int N)
{
  int idx = blockIdx.x * 256 + threadIdx.x;   // Npad * K
  int k = idx & (K - 1);
  int n = idx >> kshift;
  Wt[idx] = (n < N) ? f2bf(W[(size_t)k * N + n]) : (unsigned short)0;
}

// ---------------------------------------------------------------------------
// Time-embedding MLP chain: temb -> silu(@Wt1) -> @Wt2 -> cond=@Wcm -> silu
// ---------------------------------------------------------------------------
__global__ __launch_bounds__(256) void temb_kernel(
    const float* __restrict__ tin,
    const float* __restrict__ Wt1, const float* __restrict__ bt1,
    const float* __restrict__ Wt2, const float* __restrict__ bt2,
    const float* __restrict__ Wcm, const float* __restrict__ bcm,
    float* __restrict__ cond_act)
{
  __shared__ float s0[CD_];
  __shared__ float s1[CD_];
  const int tid = threadIdx.x;
  for (int b = 0; b < B_; ++b) {
    float tv = tin[b];
    for (int d = tid; d < CD_; d += 256) {
      int i = d & 511;
      float ang = tv * __expf(-9.210340372f * (float)i * (1.0f / 512.0f));
      s0[d] = (d < 512) ? __sinf(ang) : __cosf(ang);
    }
    __syncthreads();
    for (int j = tid; j < CD_; j += 256) {
      float acc = bt1[j];
      for (int k = 0; k < CD_; ++k) acc += s0[k] * Wt1[k * CD_ + j];
      s1[j] = acc * sigmoidf_(acc);
    }
    __syncthreads();
    for (int j = tid; j < CD_; j += 256) {
      float acc = bt2[j];
      for (int k = 0; k < CD_; ++k) acc += s1[k] * Wt2[k * CD_ + j];
      s0[j] = acc;
    }
    __syncthreads();
    for (int j = tid; j < D_; j += 256) {
      float acc = bcm[j];
      for (int k = 0; k < CD_; ++k) acc += s0[k] * Wcm[k * D_ + j];
      cond_act[b * D_ + j] = acc * sigmoidf_(acc);
    }
    __syncthreads();
  }
}

// ---------------------------------------------------------------------------
// h = x @ W_in + b_in   (K = 100, scalar)
// ---------------------------------------------------------------------------
__global__ __launch_bounds__(256) void embed_kernel(
    const float* __restrict__ x, const float* __restrict__ W,
    const float* __restrict__ bias, float* __restrict__ h)
{
  int idx = blockIdx.x * 256 + threadIdx.x;   // B*T*D
  int n = idx & (D_ - 1);
  int m = idx >> 10;
  const float* xr = x + (size_t)m * MEL_;
  float acc = bias[n];
  for (int k = 0; k < MEL_; ++k) acc += xr[k] * W[k * D_ + n];
  h[idx] = acc;
}

// ---------------------------------------------------------------------------
// ss = cond_act @ W (1024 x 2048) + b       (tiny: B=2 rows)
// ---------------------------------------------------------------------------
__global__ __launch_bounds__(256) void ada_kernel(
    const float* __restrict__ ca, const float* __restrict__ W,
    const float* __restrict__ bias, float* __restrict__ ssout)
{
  int idx = blockIdx.x * 256 + threadIdx.x;   // B * 2D
  int j = idx & (2 * D_ - 1);
  int b = idx >> 11;
  const float* c = ca + b * D_;
  float acc = bias[j];
  for (int k = 0; k < D_; ++k) acc += c[k] * W[(size_t)k * (2 * D_) + j];
  ssout[idx] = acc;
}

// ---------------------------------------------------------------------------
// Y_bf16 = LN(X) * (1+scale) + shift    (ss == nullptr -> plain LN)
// ---------------------------------------------------------------------------
__global__ __launch_bounds__(256) void ln_mod_kernel(
    const float* __restrict__ X, const float* __restrict__ ss,
    unsigned short* __restrict__ Y)
{
  __shared__ float red[256];
  const int row = blockIdx.x;
  const int b = row >> 11;            // row / T
  const int tid = threadIdx.x;
  const float* xr = X + (size_t)row * D_;
  float v0[4];
  float lsum = 0.f;
#pragma unroll
  for (int i = 0; i < 4; ++i) { v0[i] = xr[tid + i * 256]; lsum += v0[i]; }
  red[tid] = lsum; __syncthreads();
  for (int s = 128; s > 0; s >>= 1) { if (tid < s) red[tid] += red[tid + s]; __syncthreads(); }
  float mean = red[0] * (1.0f / (float)D_);
  __syncthreads();
  float lv = 0.f;
#pragma unroll
  for (int i = 0; i < 4; ++i) { float d = v0[i] - mean; lv += d * d; }
  red[tid] = lv; __syncthreads();
  for (int s = 128; s > 0; s >>= 1) { if (tid < s) red[tid] += red[tid + s]; __syncthreads(); }
  float rstd = rsqrtf(red[0] * (1.0f / (float)D_) + 1e-5f);
#pragma unroll
  for (int i = 0; i < 4; ++i) {
    int d = tid + i * 256;
    float sc = 1.0f, sh = 0.0f;
    if (ss) { sc = 1.0f + ss[b * 2 * D_ + d]; sh = ss[b * 2 * D_ + D_ + d]; }
    Y[(size_t)row * D_ + d] = f2bf((v0[i] - mean) * rstd * sc + sh);
  }
}

// ---------------------------------------------------------------------------
// WMMA GEMM: C[M,N] = A_bf16[M,K] @ Wt_bf16[Npad x K]^T + bias (+res)(gelu)
// block = 8 waves, tile 64(M) x 128(Npad); wave tile 16 x 64. Branch-free
// K-loop: 10x b128 loads + 4x v_wmma per 32-K step.
// ---------------------------------------------------------------------------
__global__ __launch_bounds__(256) void wmma_gemm(
    const unsigned short* __restrict__ A, const unsigned short* __restrict__ Wt,
    const float* __restrict__ bias, const float* __restrict__ residual,
    float* __restrict__ outF, unsigned short* __restrict__ outB,
    int M, int N, int K, int act)
{
  const int lane = threadIdx.x & 31;
  const int w = threadIdx.x >> 5;
  const int m0 = blockIdx.x * 64 + (w & 3) * 16;
  const int nbase = blockIdx.y * 128 + (w >> 2) * 64;

  v8f acc[4] = {};

  const int bn = lane & 15;
  const unsigned short* ap = A + (size_t)(m0 + bn) * K + ((lane >> 4) << 4);
  const int bko = (lane >> 4) << 4;
  const unsigned short* wp0 = Wt + (size_t)(nbase + 0  + bn) * K + bko;
  const unsigned short* wp1 = Wt + (size_t)(nbase + 16 + bn) * K + bko;
  const unsigned short* wp2 = Wt + (size_t)(nbase + 32 + bn) * K + bko;
  const unsigned short* wp3 = Wt + (size_t)(nbase + 48 + bn) * K + bko;

  for (int k0 = 0; k0 < K; k0 += 32) {
    FragU a, b0, b1, b2, b3;
    a.q[0]  = *(const uint4*)(ap + k0);
    a.q[1]  = *(const uint4*)(ap + k0 + 8);
    b0.q[0] = *(const uint4*)(wp0 + k0);
    b0.q[1] = *(const uint4*)(wp0 + k0 + 8);
    b1.q[0] = *(const uint4*)(wp1 + k0);
    b1.q[1] = *(const uint4*)(wp1 + k0 + 8);
    b2.q[0] = *(const uint4*)(wp2 + k0);
    b2.q[1] = *(const uint4*)(wp2 + k0 + 8);
    b3.q[0] = *(const uint4*)(wp3 + k0);
    b3.q[1] = *(const uint4*)(wp3 + k0 + 8);
    __builtin_prefetch(ap + k0 + 32, 0, 3);
    acc[0] = __builtin_amdgcn_wmma_f32_16x16x32_bf16(false, a.b, false, b0.b, (short)0, acc[0], false, false);
    acc[1] = __builtin_amdgcn_wmma_f32_16x16x32_bf16(false, a.b, false, b1.b, (short)0, acc[1], false, false);
    acc[2] = __builtin_amdgcn_wmma_f32_16x16x32_bf16(false, a.b, false, b2.b, (short)0, acc[2], false, false);
    acc[3] = __builtin_amdgcn_wmma_f32_16x16x32_bf16(false, a.b, false, b3.b, (short)0, acc[3], false, false);
  }

  const int rm = (lane >> 4) << 3;
#pragma unroll
  for (int j = 0; j < 4; ++j) {
    int n = nbase + j * 16 + bn;
    if (n >= N) continue;
    float bv = bias[n];
#pragma unroll
    for (int r = 0; r < 8; ++r) {
      int m = m0 + rm + r;
      float v = acc[j][r] + bv;
      if (residual) v += residual[(size_t)m * N + n];
      if (act == 1) v = 0.5f * v * (1.0f + erff(v * 0.70710678f));
      if (outF) outF[(size_t)m * N + n] = v;
      if (outB) outB[(size_t)m * N + n] = f2bf(v);
    }
  }
}

// ---------------------------------------------------------------------------
// RoPE + pack: q [B,H,T,DH], k [B,H,WIN,DH], v transposed [B,H,DH,WIN] (bf16)
// ---------------------------------------------------------------------------
__global__ __launch_bounds__(256) void rope_pack_kernel(
    const float* __restrict__ qkv,
    unsigned short* __restrict__ Qb, unsigned short* __restrict__ Kb,
    unsigned short* __restrict__ Vt)
{
  int idx = blockIdx.x * 256 + threadIdx.x;   // B*T*H*32
  int i = idx & 31;
  int hh = (idx >> 5) & (H_ - 1);
  int t = (idx >> 9) & (T_ - 1);
  int b = idx >> 20;
  const float* base = qkv + (size_t)(b * T_ + t) * (3 * D_);
  float freq = __expf(-9.210340372f * (float)(2 * i) * (1.0f / (float)DH_));
  float s, c;
  __sincosf((float)t * freq, &s, &c);
  {
    const float* qr = base + hh * DH_;
    float x1 = qr[2 * i], x2 = qr[2 * i + 1];
    size_t off = ((size_t)(b * H_ + hh) * T_ + t) * DH_ + 2 * i;
    Qb[off] = f2bf(x1 * c - x2 * s);
    Qb[off + 1] = f2bf(x1 * s + x2 * c);
  }
  if (t >= T_ - WIN_) {
    int wp = t - (T_ - WIN_);
    int bh = b * H_ + hh;
    size_t koff = ((size_t)bh * WIN_ + wp) * DH_ + 2 * i;
    const float* kr = base + D_ + hh * DH_;
    float x1 = kr[2 * i], x2 = kr[2 * i + 1];
    Kb[koff] = f2bf(x1 * c - x2 * s);
    Kb[koff + 1] = f2bf(x1 * s + x2 * c);
    const float* vr = base + 2 * D_ + hh * DH_;
    Vt[((size_t)bh * DH_ + 2 * i) * WIN_ + wp]     = f2bf(vr[2 * i]);
    Vt[((size_t)bh * DH_ + 2 * i + 1) * WIN_ + wp] = f2bf(vr[2 * i + 1]);
  }
}

// ---------------------------------------------------------------------------
// Flash-style windowed attention. One wave per (b,h,16-query tile).
// ---------------------------------------------------------------------------
__global__ __launch_bounds__(256) void attn_kernel(
    const unsigned short* __restrict__ Q, const unsigned short* __restrict__ Km,
    const unsigned short* __restrict__ Vt, unsigned short* __restrict__ O)
{
  __shared__ unsigned short plds[8 * 512];
  const int lane = threadIdx.x & 31;
  const int w = threadIdx.x >> 5;
  const int g = blockIdx.x * 8 + w;
  const int qt = g & 127;           // T/16 tiles
  const int bh = g >> 7;
  const int m0 = qt * 16;
  const int b = bh >> 4;
  const int hh = bh & 15;

  const unsigned short* Qp = Q + (size_t)bh * T_ * DH_;
  const unsigned short* Kp = Km + (size_t)bh * WIN_ * DH_;
  const unsigned short* Vp = Vt + (size_t)bh * DH_ * WIN_;

  const int half = lane >> 4;
  const int col = lane & 15;

  FragU qa0, qa1;
  {
    const unsigned short* qp = Qp + (size_t)(m0 + col) * DH_ + (half << 4);
    qa0.q[0] = *(const uint4*)(qp);
    qa0.q[1] = *(const uint4*)(qp + 8);
    qa1.q[0] = *(const uint4*)(qp + 32);
    qa1.q[1] = *(const uint4*)(qp + 40);
  }

  v8f o[4] = {};
  float rmax[8], rsum[8];
#pragma unroll
  for (int r = 0; r < 8; ++r) { rmax[r] = -3.0e38f; rsum[r] = 0.0f; }

  const int jend = (WIN_ < m0 + 16) ? WIN_ : (m0 + 16);
  unsigned short* pl = plds + w * 512;

  for (int j0 = 0; j0 < jend; j0 += 32) {
    v8f s0 = {}, s1 = {};
    {
      FragU k0f, k1f;
      const unsigned short* kp = Kp + (size_t)(j0 + col) * DH_ + (half << 4);
      k0f.q[0] = *(const uint4*)(kp);
      k0f.q[1] = *(const uint4*)(kp + 8);
      k1f.q[0] = *(const uint4*)(kp + 32);
      k1f.q[1] = *(const uint4*)(kp + 40);
      s0 = __builtin_amdgcn_wmma_f32_16x16x32_bf16(false, qa0.b, false, k0f.b, (short)0, s0, false, false);
      s0 = __builtin_amdgcn_wmma_f32_16x16x32_bf16(false, qa1.b, false, k1f.b, (short)0, s0, false, false);
    }
    {
      FragU k0f, k1f;
      const unsigned short* kp = Kp + (size_t)(j0 + 16 + col) * DH_ + (half << 4);
      k0f.q[0] = *(const uint4*)(kp);
      k0f.q[1] = *(const uint4*)(kp + 8);
      k1f.q[0] = *(const uint4*)(kp + 32);
      k1f.q[1] = *(const uint4*)(kp + 40);
      s1 = __builtin_amdgcn_wmma_f32_16x16x32_bf16(false, qa0.b, false, k0f.b, (short)0, s1, false, false);
      s1 = __builtin_amdgcn_wmma_f32_16x16x32_bf16(false, qa1.b, false, k1f.b, (short)0, s1, false, false);
    }
    // mask + online softmax
#pragma unroll
    for (int r = 0; r < 8; ++r) {
      int qpos = m0 + r + (half << 3);
      float a = s0[r] * 0.125f; if (j0 + col > qpos) a = -3.0e38f;
      float c = s1[r] * 0.125f; if (j0 + 16 + col > qpos) c = -3.0e38f;
      float mx = fmaxf(a, c);
#pragma unroll
      for (int off = 1; off < 16; off <<= 1) mx = fmaxf(mx, __shfl_xor(mx, off, 32));
      float newm = fmaxf(rmax[r], mx);
      float corr = __expf(rmax[r] - newm);
      rsum[r] *= corr;
      o[0][r] *= corr; o[1][r] *= corr; o[2][r] *= corr; o[3][r] *= corr;
      float p0 = __expf(a - newm);
      float p1 = __expf(c - newm);
      float ps = p0 + p1;
#pragma unroll
      for (int off = 1; off < 16; off <<= 1) ps += __shfl_xor(ps, off, 32);
      rsum[r] += ps;
      rmax[r] = newm;
      s0[r] = p0; s1[r] = p1;
    }
    // P -> LDS (C-layout) -> reload as A-fragment (16x32)
#pragma unroll
    for (int r = 0; r < 8; ++r) {
      int m = r + (half << 3);
      pl[m * 32 + col] = f2bf(s0[r]);
      pl[m * 32 + 16 + col] = f2bf(s1[r]);
    }
    asm volatile("s_wait_dscnt 0" ::: "memory");
    FragU pf;
    {
      const unsigned short* pp = pl + col * 32 + (half << 4);
      pf.q[0] = *(const uint4*)(pp);
      pf.q[1] = *(const uint4*)(pp + 8);
    }
    // O += P @ V   (V pre-transposed: rows = dh)
    const int jb = j0 + (half << 4);
#pragma unroll
    for (int nn = 0; nn < 4; ++nn) {
      FragU vf;
      const unsigned short* vp = Vp + (size_t)((nn << 4) + col) * WIN_ + jb;
      vf.q[0] = *(const uint4*)(vp);
      vf.q[1] = *(const uint4*)(vp + 8);
      o[nn] = __builtin_amdgcn_wmma_f32_16x16x32_bf16(false, pf.b, false, vf.b, (short)0, o[nn], false, false);
    }
  }

  unsigned short* Ob = O + (size_t)b * T_ * D_ + hh * DH_;
#pragma unroll
  for (int r = 0; r < 8; ++r) {
    int m = r + (half << 3);
    float inv = 1.0f / rsum[r];
    unsigned short* orow = Ob + (size_t)(m0 + m) * D_;
    orow[0 + col]  = f2bf(o[0][r] * inv);
    orow[16 + col] = f2bf(o[1][r] * inv);
    orow[32 + col] = f2bf(o[2][r] * inv);
    orow[48 + col] = f2bf(o[3][r] * inv);
  }
}

// ---------------------------------------------------------------------------
extern "C" void kernel_launch(void* const* d_in, const int* in_sizes, int n_in,
                              void* d_out, int out_size, void* d_ws, size_t ws_size,
                              hipStream_t stream)
{
  const float* x      = (const float*)d_in[0];
  const float* t      = (const float*)d_in[1];
  const float* W_in   = (const float*)d_in[2];
  const float* b_in   = (const float*)d_in[3];
  const float* Wt1    = (const float*)d_in[4];
  const float* bt1    = (const float*)d_in[5];
  const float* Wt2    = (const float*)d_in[6];
  const float* bt2    = (const float*)d_in[7];
  const float* Wcm    = (const float*)d_in[8];
  const float* bcm    = (const float*)d_in[9];
  const float* ada1_W = (const float*)d_in[10];
  const float* ada1_b = (const float*)d_in[11];
  const float* qkv_W  = (const float*)d_in[12];
  const float* qkv_b  = (const float*)d_in[13];
  const float* atW    = (const float*)d_in[14];
  const float* atb    = (const float*)d_in[15];
  const float* ada2_W = (const float*)d_in[16];
  const float* ada2_b = (const float*)d_in[17];
  const float* m1W    = (const float*)d_in[18];
  const float* m1b    = (const float*)d_in[19];
  const float* m2W    = (const float*)d_in[20];
  const float* m2b    = (const float*)d_in[21];
  const float* W_out  = (const float*)d_in[22];
  const float* b_out  = (const float*)d_in[23];

  char* wsp = (char*)d_ws;
  auto alloc = [&](size_t bytes) -> void* {
    void* p = (void*)wsp;
    wsp += (bytes + 255) & ~(size_t)255;
    return p;
  };
  const size_t MT = (size_t)B_ * T_;   // 4096 rows

  float*          cond_act = (float*)alloc((size_t)B_ * D_ * 4);
  float*          ssbuf    = (float*)alloc((size_t)B_ * 2 * D_ * 4);
  float*          h        = (float*)alloc(MT * D_ * 4);
  unsigned short* hn       = (unsigned short*)alloc(MT * D_ * 2);
  float*          qkvbuf   = (float*)alloc(MT * 3 * D_ * 4);
  unsigned short* qb       = (unsigned short*)alloc((size_t)B_ * H_ * T_ * DH_ * 2);
  unsigned short* kb       = (unsigned short*)alloc((size_t)B_ * H_ * WIN_ * DH_ * 2);
  unsigned short* vt       = (unsigned short*)alloc((size_t)B_ * H_ * DH_ * WIN_ * 2);
  unsigned short* ob       = (unsigned short*)alloc(MT * D_ * 2);
  unsigned short* gb       = (unsigned short*)alloc(MT * FF_ * 2);
  unsigned short* wtbuf    = (unsigned short*)alloc((size_t)FF_ * D_ * 2);  // 8MB, reused

  temb_kernel<<<1, 256, 0, stream>>>(t, Wt1, bt1, Wt2, bt2, Wcm, bcm, cond_act);
  embed_kernel<<<(int)(MT * D_ / 256), 256, 0, stream>>>(x, W_in, b_in, h);

  for (int l = 0; l < L_; ++l) {
    // --- attention block ---
    ada_kernel<<<(B_ * 2 * D_) / 256, 256, 0, stream>>>(
        cond_act, ada1_W + (size_t)l * D_ * 2 * D_, ada1_b + (size_t)l * 2 * D_, ssbuf);
    ln_mod_kernel<<<(int)MT, 256, 0, stream>>>(h, ssbuf, hn);
    wt_convert<<<(3 * D_ * D_) / 256, 256, 0, stream>>>(
        qkv_W + (size_t)l * D_ * 3 * D_, wtbuf, D_, 10, 3 * D_);
    wmma_gemm<<<dim3(64, 24), 256, 0, stream>>>(
        hn, wtbuf, qkv_b + (size_t)l * 3 * D_,
        nullptr, qkvbuf, nullptr, (int)MT, 3 * D_, D_, 0);
    rope_pack_kernel<<<(B_ * T_ * H_ * 32) / 256, 256, 0, stream>>>(qkvbuf, qb, kb, vt);
    attn_kernel<<<(B_ * H_ * (T_ / 16)) / 8, 256, 0, stream>>>(qb, kb, vt, ob);
    wt_convert<<<(D_ * D_) / 256, 256, 0, stream>>>(
        atW + (size_t)l * D_ * D_, wtbuf, D_, 10, D_);
    wmma_gemm<<<dim3(64, 8), 256, 0, stream>>>(
        ob, wtbuf, atb + (size_t)l * D_,
        h, h, nullptr, (int)MT, D_, D_, 0);
    // --- MLP block ---
    ada_kernel<<<(B_ * 2 * D_) / 256, 256, 0, stream>>>(
        cond_act, ada2_W + (size_t)l * D_ * 2 * D_, ada2_b + (size_t)l * 2 * D_, ssbuf);
    ln_mod_kernel<<<(int)MT, 256, 0, stream>>>(h, ssbuf, hn);
    wt_convert<<<(FF_ * D_) / 256, 256, 0, stream>>>(
        m1W + (size_t)l * D_ * FF_, wtbuf, D_, 10, FF_);
    wmma_gemm<<<dim3(64, 32), 256, 0, stream>>>(
        hn, wtbuf, m1b + (size_t)l * FF_,
        nullptr, nullptr, gb, (int)MT, FF_, D_, 1);
    wt_convert<<<(D_ * FF_) / 256, 256, 0, stream>>>(
        m2W + (size_t)l * FF_ * D_, wtbuf, FF_, 12, D_);
    wmma_gemm<<<dim3(64, 8), 256, 0, stream>>>(
        gb, wtbuf, m2b + (size_t)l * D_,
        h, h, nullptr, (int)MT, D_, FF_, 0);
  }

  ln_mod_kernel<<<(int)MT, 256, 0, stream>>>(h, nullptr, hn);
  wt_convert<<<(128 * D_) / 256, 256, 0, stream>>>(W_out, wtbuf, D_, 10, MEL_);
  wmma_gemm<<<dim3(64, 1), 256, 0, stream>>>(
      hn, wtbuf, b_out, nullptr, (float*)d_out, nullptr, (int)MT, MEL_, D_, 0);
}
